// MultiHeadCrossAttention_52845277610468
// MI455X (gfx1250) — matrix operand order repaired and checked by reading, not compiled, not run on hardware
//
#include <hip/hip_runtime.h>
#include <hip/hip_bf16.h>
#include <stdint.h>

typedef __attribute__((ext_vector_type(16))) _Float16 v16h;
typedef __attribute__((ext_vector_type(8)))  _Float16 v8h;
typedef __attribute__((ext_vector_type(8)))  float    v8f;
typedef __attribute__((ext_vector_type(4)))  float    v4f;

#define NB    8
#define LQ    512
#define LKV   2048
#define QDIM  768
#define HDIM  1024
#define NH    16
#define DH    64
#define ATT_SCALE 0.125f      // HEAD_DIM^-0.5 = 64^-0.5
#define NEG_BIG  (-3.0e38f)
#define SSTR  36              // S-transpose LDS row stride in floats
#define KSTR  72              // K stage row stride in halves (144 B rows)
#define VSTR  40              // V stage row stride in halves (80 B rows)

// gfx1250 async global->LDS copy (ASYNCcnt-tracked DMA path)
__device__ __forceinline__ void async_ld_b128(unsigned lds_addr, unsigned voff,
                                              const void* sbase) {
    asm volatile("global_load_async_to_lds_b128 %0, %1, %2"
                 :: "v"(lds_addr), "v"(voff), "s"(sbase) : "memory");
}
__device__ __forceinline__ void wait_asynccnt0() {
    asm volatile("s_wait_asynccnt 0x0" ::: "memory");
}

// ---------------------------------------------------------------------------
// Kernel 0: transpose + convert weights to f16, [K][N] -> [N][K]
// ---------------------------------------------------------------------------
__global__ void prep_weights(const float* __restrict__ Wq, const float* __restrict__ Wk,
                             const float* __restrict__ Wv, const float* __restrict__ Wo,
                             _Float16* __restrict__ WqT, _Float16* __restrict__ WkT,
                             _Float16* __restrict__ WvT, _Float16* __restrict__ WoT) {
    const int stride = gridDim.x * blockDim.x;
    const int tid = blockIdx.x * blockDim.x + threadIdx.x;
    for (int i = tid; i < QDIM * HDIM; i += stride) {
        int k = i / HDIM, n = i - k * HDIM;
        WqT[n * QDIM + k] = (_Float16)Wq[i];
        WkT[n * QDIM + k] = (_Float16)Wk[i];
        WvT[n * QDIM + k] = (_Float16)Wv[i];
    }
    for (int i = tid; i < HDIM * HDIM; i += stride) {
        int k = i / HDIM, n = i - k * HDIM;
        WoT[n * HDIM + k] = (_Float16)Wo[i];
    }
}

// ---------------------------------------------------------------------------
// Kernel 1: projection GEMM  out = (X @ W + b) * scale
//   MODE 0: Q -> [b,h,q,d]   MODE 1: K -> [b,h,kv,d]   MODE 2: V^T -> [b,h,d,kv]
// ---------------------------------------------------------------------------
template <int MODE>
__global__ __launch_bounds__(256)
void proj_kernel(const float* __restrict__ X, const _Float16* __restrict__ WT,
                 const float* __restrict__ bias, _Float16* __restrict__ out,
                 int L, float scale) {
    const int lane = threadIdx.x & 31;
    const int hl   = lane >> 4;
    const int rr   = lane & 15;
    const int w    = threadIdx.x >> 5;
    const int tile = blockIdx.x * 8 + w;
    const int ntn  = HDIM / 16;
    const int mt   = tile / ntn;
    const int nt   = tile - mt * ntn;
    const int m0   = mt * 16;
    const int n0   = nt * 16;

    const float*    arow = X  + (size_t)(m0 + rr) * QDIM;
    const _Float16* brow = WT + (size_t)(n0 + rr) * QDIM;

    v8f acc = {};
    for (int k0 = 0; k0 < QDIM; k0 += 32) {
        v4f a0 = *(const v4f*)(arow + k0 + 8 * hl);
        v4f a1 = *(const v4f*)(arow + k0 + 8 * hl + 4);
        v4f a2 = *(const v4f*)(arow + k0 + 8 * hl + 16);
        v4f a3 = *(const v4f*)(arow + k0 + 8 * hl + 20);
        v8h b0 = *(const v8h*)(brow + k0 + 8 * hl);
        v8h b1 = *(const v8h*)(brow + k0 + 8 * hl + 16);
        v16h a, b;
#pragma unroll
        for (int i = 0; i < 4; ++i) {
            a[i]      = (_Float16)a0[i];
            a[4 + i]  = (_Float16)a1[i];
            a[8 + i]  = (_Float16)a2[i];
            a[12 + i] = (_Float16)a3[i];
        }
#pragma unroll
        for (int i = 0; i < 8; ++i) { b[i] = b0[i]; b[8 + i] = b1[i]; }
        acc = __builtin_amdgcn_wmma_f32_16x16x32_f16(false, a, false, b, (short)0, acc,
                                                     false, false);
    }

    const float bn = bias[n0 + rr];
    const int nglob = n0 + rr;
    const int h = nglob >> 6;
    const int d = nglob & 63;
#pragma unroll
    for (int r = 0; r < 8; ++r) {
        int m = m0 + r + 8 * hl;
        int bidx = m / L;
        int l = m - bidx * L;
        float val = (acc[r] + bn) * scale;
        size_t addr;
        if (MODE == 0)      addr = (((size_t)bidx * NH + h) * LQ  + l) * DH + d;
        else if (MODE == 1) addr = (((size_t)bidx * NH + h) * LKV + l) * DH + d;
        else                addr = (((size_t)bidx * NH + h) * DH  + d) * LKV + l;
        out[addr] = (_Float16)val;
    }
}

// ---------------------------------------------------------------------------
// Kernel 2: flash attention with async-LDS staging of K/V stripes.
//   grid = B*H*4 blocks, 8 waves/block, wave = 16 q-rows x Dh=64.
//   Each 32-kv stripe of K (4KB) and V^T (4KB) is DMA'd to LDS once per block
//   (global_load_async_to_lds_b128, double-buffered), then all 8 waves read
//   their WMMA B-fragments from LDS.
// ---------------------------------------------------------------------------
__global__ __launch_bounds__(256)
void attn_kernel(const _Float16* __restrict__ Qh, const _Float16* __restrict__ Kh,
                 const _Float16* __restrict__ VTh, const unsigned char* __restrict__ kv_mask,
                 _Float16* __restrict__ Oh) {
    __shared__ __align__(16) float    slds[8 * 16 * SSTR];   // 18 KB: S transpose
    __shared__ __align__(16) _Float16 kstage[2][32 * KSTR];  //  9 KB: K stripes
    __shared__ __align__(16) _Float16 vstage[2][64 * VSTR];  // 10 KB: V stripes

    const int tid  = threadIdx.x;
    const int lane = tid & 31;
    const int hl   = lane >> 4;
    const int rr   = lane & 15;
    const int w    = tid >> 5;
    const int bh   = blockIdx.x >> 2;     // b*NH + h
    const int qb   = blockIdx.x & 3;
    const int b    = bh >> 4;
    const int h    = bh & 15;
    const int q0   = qb * 128 + w * 16;

    const _Float16* kbase = Kh  + (size_t)bh * LKV * DH;
    const _Float16* vbase = VTh + (size_t)bh * DH * LKV;
    const unsigned char* mbase = kv_mask + (size_t)b * LKV;

    // Per-thread staging assignment: one b128 of K and one of V per stripe.
    const int krow = tid >> 3, kchk = tid & 7;      // K stripe: 32 rows x 128 B
    const int vrow = tid >> 2, vchk = tid & 3;      // V stripe: 64 rows x 64 B
    const unsigned klds0 = (unsigned)(uintptr_t)&kstage[0][0] + krow * (KSTR * 2) + kchk * 16;
    const unsigned klds1 = (unsigned)(uintptr_t)&kstage[1][0] + krow * (KSTR * 2) + kchk * 16;
    const unsigned vlds0 = (unsigned)(uintptr_t)&vstage[0][0] + vrow * (VSTR * 2) + vchk * 16;
    const unsigned vlds1 = (unsigned)(uintptr_t)&vstage[1][0] + vrow * (VSTR * 2) + vchk * 16;
    const unsigned kvoffb = (unsigned)(krow * (DH * 2) + kchk * 16);     // + kv0*128
    const unsigned vvoffb = (unsigned)(vrow * (LKV * 2) + vchk * 16);    // + kv0*2

    // Q A-fragments over d: aq0 = d[0,32), aq1 = d[32,64). Q pre-scaled by 1/8.
    const _Float16* qrow = Qh + (((size_t)bh) * LQ + q0 + rr) * DH;
    v16h aq0, aq1;
    {
        v8h t0 = *(const v8h*)(qrow + 8 * hl);
        v8h t1 = *(const v8h*)(qrow + 8 * hl + 16);
        v8h t2 = *(const v8h*)(qrow + 32 + 8 * hl);
        v8h t3 = *(const v8h*)(qrow + 32 + 8 * hl + 16);
#pragma unroll
        for (int i = 0; i < 8; ++i) {
            aq0[i] = t0[i]; aq0[8 + i] = t1[i];
            aq1[i] = t2[i]; aq1[8 + i] = t3[i];
        }
    }

    v8f o[4] = {{}, {}, {}, {}};
    float mval = NEG_BIG;   // running max of row rr (dup in lanes rr, rr+16)
    float lval = 0.f;       // running sum of row rr
    float* sw = slds + w * 16 * SSTR;

    // Prefetch stripe 0 into buffer 0.
    async_ld_b128(klds0, kvoffb, kbase);
    async_ld_b128(vlds0, vvoffb, vbase);

    for (int kv0 = 0; kv0 < LKV; kv0 += 32) {
        const int bufi = (kv0 >> 5) & 1;
        wait_asynccnt0();
        __syncthreads();          // stripe ready; prev iteration's LDS reads done
        if (kv0 + 32 < LKV) {     // prefetch next stripe into the other buffer
            unsigned kv = (unsigned)(kv0 + 32);
            async_ld_b128(bufi ? klds0 : klds1, kvoffb + kv * (DH * 2), kbase);
            async_ld_b128(bufi ? vlds0 : vlds1, vvoffb + kv * 2, vbase);
        }
        const _Float16* kst = &kstage[bufi][0];
        const _Float16* vst = &vstage[bufi][0];

        // ---- S = Q K^T over a 16x32 kv stripe (two 16x16 C tiles) ----
        v8f s[2];
        float madd[2];
#pragma unroll
        for (int t = 0; t < 2; ++t) {
            const _Float16* krw = kst + (t * 16 + rr) * KSTR;
            v16h bk0, bk1;
            {
                v8h u0 = *(const v8h*)(krw + 8 * hl);
                v8h u1 = *(const v8h*)(krw + 8 * hl + 16);
                v8h u2 = *(const v8h*)(krw + 32 + 8 * hl);
                v8h u3 = *(const v8h*)(krw + 32 + 8 * hl + 16);
#pragma unroll
                for (int i = 0; i < 8; ++i) {
                    bk0[i] = u0[i]; bk0[8 + i] = u1[i];
                    bk1[i] = u2[i]; bk1[8 + i] = u3[i];
                }
            }
            v8f z = {};
            z = __builtin_amdgcn_wmma_f32_16x16x32_f16(false, aq0, false, bk0, (short)0, z,
                                                       false, false);
            z = __builtin_amdgcn_wmma_f32_16x16x32_f16(false, aq1, false, bk1, (short)0, z,
                                                       false, false);
            s[t] = z;
            // branchless additive mask for this lane's kv column
            madd[t] = mbase[kv0 + t * 16 + rr] ? 0.f : NEG_BIG;
        }
        // ---- transpose S (+mask) through LDS: C layout -> A layout (f32) ----
#pragma unroll
        for (int t = 0; t < 2; ++t)
#pragma unroll
            for (int r = 0; r < 8; ++r)
                sw[(r + 8 * hl) * SSTR + t * 16 + rr] = s[t][r] + madd[t];
        __syncthreads();
        // lane (rr, hl) now reads its 16 A-fragment scores of row rr
        v4f c0 = *(const v4f*)(sw + rr * SSTR + 8 * hl);
        v4f c1 = *(const v4f*)(sw + rr * SSTR + 8 * hl + 4);
        v4f c2 = *(const v4f*)(sw + rr * SSTR + 16 + 8 * hl);
        v4f c3 = *(const v4f*)(sw + rr * SSTR + 16 + 8 * hl + 4);
        // ---- row max: local over 16 + one xor-16 shuffle ----
        float vmax = c0[0];
#pragma unroll
        for (int i = 1; i < 4; ++i) vmax = fmaxf(vmax, c0[i]);
#pragma unroll
        for (int i = 0; i < 4; ++i) {
            vmax = fmaxf(vmax, c1[i]);
            vmax = fmaxf(vmax, c2[i]);
            vmax = fmaxf(vmax, c3[i]);
        }
        vmax = fmaxf(vmax, __shfl_xor(vmax, 16, 32));
        float mnew = fmaxf(mval, vmax);
        float corr = __expf(mval - mnew);
        mval = mnew;
        // ---- p = exp(s - m), packed straight into the PV A-fragment ----
        v16h pa;
        float rs = 0.f;
#pragma unroll
        for (int i = 0; i < 4; ++i) {
            float p0 = __expf(c0[i] - mnew);
            float p1 = __expf(c1[i] - mnew);
            float p2 = __expf(c2[i] - mnew);
            float p3 = __expf(c3[i] - mnew);
            pa[i]      = (_Float16)p0;
            pa[4 + i]  = (_Float16)p1;
            pa[8 + i]  = (_Float16)p2;
            pa[12 + i] = (_Float16)p3;
            rs += p0 + p1 + p2 + p3;
        }
        rs += __shfl_xor(rs, 16, 32);
        lval = lval * corr + rs;
        // ---- rescale O: broadcast per-row corr (row r+8*hl lives in lane r+8*hl)
#pragma unroll
        for (int r = 0; r < 8; ++r) {
            float c = __shfl(corr, 8 * hl + r, 32);
            o[0][r] *= c; o[1][r] *= c; o[2][r] *= c; o[3][r] *= c;
        }
        // ---- O += P V (V stripe in LDS, contiguous along kv) ----
#pragma unroll
        for (int dt = 0; dt < 4; ++dt) {
            const _Float16* vrw = vst + (dt * 16 + rr) * VSTR + 8 * hl;
            v8h u0 = *(const v8h*)(vrw);
            v8h u1 = *(const v8h*)(vrw + 16);
            v16h bv;
#pragma unroll
            for (int i = 0; i < 8; ++i) { bv[i] = u0[i]; bv[8 + i] = u1[i]; }
            o[dt] = __builtin_amdgcn_wmma_f32_16x16x32_f16(false, pa, false, bv, (short)0,
                                                           o[dt], false, false);
        }
    }

    // ---- finalize: divide by row sum, store O as f16 [b, q, h*64+d] ----
#pragma unroll
    for (int r = 0; r < 8; ++r) {
        float lr = __shfl(lval, 8 * hl + r, 32);
        float inv = 1.0f / lr;
        int q = q0 + r + 8 * hl;
#pragma unroll
        for (int dt = 0; dt < 4; ++dt) {
            int d = dt * 16 + rr;
            Oh[((size_t)b * LQ + q) * HDIM + h * DH + d] = (_Float16)(o[dt][r] * inv);
        }
    }
}

// ---------------------------------------------------------------------------
// Kernel 3: out = O @ Wo + bo   (f16 x f16 -> f32), M=4096, N=K=1024
// ---------------------------------------------------------------------------
__global__ __launch_bounds__(256)
void out_gemm(const _Float16* __restrict__ Oh, const _Float16* __restrict__ WoT,
              const float* __restrict__ bo, float* __restrict__ out) {
    const int lane = threadIdx.x & 31;
    const int hl   = lane >> 4;
    const int rr   = lane & 15;
    const int w    = threadIdx.x >> 5;
    const int tile = blockIdx.x * 8 + w;
    const int ntn  = HDIM / 16;
    const int mt   = tile / ntn;
    const int nt   = tile - mt * ntn;
    const int m0   = mt * 16;
    const int n0   = nt * 16;

    const _Float16* arow = Oh  + (size_t)(m0 + rr) * HDIM;
    const _Float16* brow = WoT + (size_t)(n0 + rr) * HDIM;

    v8f acc = {};
    for (int k0 = 0; k0 < HDIM; k0 += 32) {
        v8h a0 = *(const v8h*)(arow + k0 + 8 * hl);
        v8h a1 = *(const v8h*)(arow + k0 + 8 * hl + 16);
        v8h b0 = *(const v8h*)(brow + k0 + 8 * hl);
        v8h b1 = *(const v8h*)(brow + k0 + 8 * hl + 16);
        v16h a, b;
#pragma unroll
        for (int i = 0; i < 8; ++i) {
            a[i] = a0[i]; a[8 + i] = a1[i];
            b[i] = b0[i]; b[8 + i] = b1[i];
        }
        acc = __builtin_amdgcn_wmma_f32_16x16x32_f16(false, a, false, b, (short)0, acc,
                                                     false, false);
    }
    const float bn = bo[n0 + rr];
#pragma unroll
    for (int r = 0; r < 8; ++r)
        out[(size_t)(m0 + r + 8 * hl) * HDIM + n0 + rr] = acc[r] + bn;
}

// ---------------------------------------------------------------------------
extern "C" void kernel_launch(void* const* d_in, const int* in_sizes, int n_in,
                              void* d_out, int out_size, void* d_ws, size_t ws_size,
                              hipStream_t stream) {
    (void)in_sizes; (void)n_in; (void)out_size; (void)ws_size;

    const float* query      = (const float*)d_in[0];
    const float* key_value  = (const float*)d_in[1];
    const unsigned char* kv_mask = (const unsigned char*)d_in[2];
    const float* Wq = (const float*)d_in[3];
    const float* bq = (const float*)d_in[4];
    const float* Wk = (const float*)d_in[5];
    const float* bk = (const float*)d_in[6];
    const float* Wv = (const float*)d_in[7];
    const float* bv = (const float*)d_in[8];
    const float* Wo = (const float*)d_in[9];
    const float* bo = (const float*)d_in[10];
    float* out = (float*)d_out;

    char* ws = (char*)d_ws;
    size_t off = 0;
    auto alloc = [&](size_t bytes) -> void* {
        void* p = ws + off;
        off = (off + bytes + 255) & ~(size_t)255;
        return p;
    };
    _Float16* WqT = (_Float16*)alloc((size_t)QDIM * HDIM * 2);
    _Float16* WkT = (_Float16*)alloc((size_t)QDIM * HDIM * 2);
    _Float16* WvT = (_Float16*)alloc((size_t)QDIM * HDIM * 2);
    _Float16* WoT = (_Float16*)alloc((size_t)HDIM * HDIM * 2);
    _Float16* Qh  = (_Float16*)alloc((size_t)NB * NH * LQ  * DH * 2);
    _Float16* Kh  = (_Float16*)alloc((size_t)NB * NH * LKV * DH * 2);
    _Float16* VTh = (_Float16*)alloc((size_t)NB * NH * DH * LKV * 2);
    _Float16* Oh  = (_Float16*)alloc((size_t)NB * LQ * HDIM * 2);

    prep_weights<<<1024, 256, 0, stream>>>(Wq, Wk, Wv, Wo, WqT, WkT, WvT, WoT);

    proj_kernel<0><<<(4096 / 16) * (HDIM / 16) / 8, 256, 0, stream>>>(
        query, WqT, bq, Qh, LQ, ATT_SCALE);
    proj_kernel<1><<<(16384 / 16) * (HDIM / 16) / 8, 256, 0, stream>>>(
        key_value, WkT, bk, Kh, LKV, 1.0f);
    proj_kernel<2><<<(16384 / 16) * (HDIM / 16) / 8, 256, 0, stream>>>(
        key_value, WvT, bv, VTh, LKV, 1.0f);

    attn_kernel<<<NB * NH * 4, 256, 0, stream>>>(Qh, Kh, VTh, kv_mask, Oh);

    out_gemm<<<(4096 / 16) * (HDIM / 16) / 8, 256, 0, stream>>>(Oh, WoT, bo, out);
}
